// Projector_79577154060539
// MI455X (gfx1250) — matrix operand order repaired and checked by reading, not compiled, not run on hardware
//
#include <hip/hip_runtime.h>
#include <hip/hip_bf16.h>

typedef __attribute__((ext_vector_type(16))) __bf16 v16bf;
typedef __attribute__((ext_vector_type(8)))  __bf16 v8bf;
typedef __attribute__((ext_vector_type(8)))  float  v8f;

#define DSZ 64
#define DD  4096          // 64*64
#define BLK 128           // 4 waves of 32

#define WMMA(a, b, c) \
  __builtin_amdgcn_wmma_f32_16x16x32_bf16(false, (a), false, (b), (short)0, (c), false, false)

// ---- WMMA fragment helpers (layouts per CDNA5 ISA §7.12.2) ----
// All loads/stores are contiguous, 16/32-byte aligned vector LDS accesses.

// A fragment 16x32 (MxK) from row-major M: lane half h, m = lane&15.
// Elements = M[row][k0 + h*8 .. +7] ++ M[row][k0 + 16 + h*8 .. +7]  (2x 16B)
__device__ __forceinline__ v16bf frag_a(const __bf16* M, int row0, int k0, int lane) {
  const int half = lane >> 4; const int row = row0 + (lane & 15);
  const __bf16* p = &M[row * DSZ + k0 + half * 8];
  const v8bf lo = *(const v8bf*)p;
  const v8bf hi = *(const v8bf*)(p + 16);
  v16bf f;
#pragma unroll
  for (int i = 0; i < 8; ++i) { f[i] = lo[i]; f[8 + i] = hi[i]; }
  return f;
}

// B fragment 32x16 (KxN) from TRANSPOSED storage ST[n][k] = B[k][n]:
// lane's 16 elements = ST[col][k0 + half*8 .. +7] ++ ST[col][k0 + 16 + half*8 .. +7]  (2x 16B)
__device__ __forceinline__ v16bf frag_bT(const __bf16* ST, int k0, int n0, int lane) {
  const int half = lane >> 4; const int col = n0 + (lane & 15);
  const __bf16* p = &ST[col * DSZ + k0 + half * 8];
  const v8bf lo = *(const v8bf*)p;
  const v8bf hi = *(const v8bf*)(p + 16);
  v16bf f;
#pragma unroll
  for (int i = 0; i < 8; ++i) { f[i] = lo[i]; f[8 + i] = hi[i]; }
  return f;
}

// D tile 16x16 stored TRANSPOSED: lane's 8 rows are contiguous in ST[col][...]
__device__ __forceinline__ void store_T_bf16(__bf16* ST, int row0, int n0, int lane, v8f c) {
  const int half = lane >> 4; const int col = n0 + (lane & 15);
  v8bf w;
#pragma unroll
  for (int g = 0; g < 8; ++g) w[g] = (__bf16)c[g];
  *(v8bf*)&ST[col * DSZ + row0 + half * 8] = w;
}
__device__ __forceinline__ void store_T_f32(float* ST, int row0, int n0, int lane, v8f c) {
  const int half = lane >> 4; const int col = n0 + (lane & 15);
  *(v8f*)&ST[col * DSZ + row0 + half * 8] = c;
}

// ---- block reductions ----
__device__ __forceinline__ float red_max(float v, float* red) {
  const int t = threadIdx.x; red[t] = v; __syncthreads();
#pragma unroll
  for (int s = 64; s > 0; s >>= 1) {
    if (t < s) red[t] = fmaxf(red[t], red[t + s]);
    __syncthreads();
  }
  float r = red[0]; __syncthreads(); return r;
}
__device__ __forceinline__ float red_sum(float v, float* red) {
  const int t = threadIdx.x; red[t] = v; __syncthreads();
#pragma unroll
  for (int s = 64; s > 0; s >>= 1) {
    if (t < s) red[t] += red[t + s];
    __syncthreads();
  }
  float r = red[0]; __syncthreads(); return r;
}

// =====================================================================
// Kernel 1: batch-independent _log_norm -> d_ws[0]
//   Uses Al symmetry:  U_c = Al @ M_c ;  Al' = sum_c M_c^T @ U_c
// =====================================================================
__global__ __launch_bounds__(BLK) void lognorm_kernel(
    const float* __restrict__ Mi_all,   // [127,64,64,2]
    const float* __restrict__ Mo_all,   // [129,64,64,2,2]
    float* __restrict__ lognorm) {
  __shared__ __align__(32) float  s_Al[DD];
  __shared__ __align__(32) __bf16 s_Albf[DD];
  __shared__ __align__(32) __bf16 s_Mt[4][DD];   // ST_M[c][k][d] = M_c[d][k]
  __shared__ __align__(32) __bf16 s_Ut[4][DD];   // ST_U[c][n][k] = U_c[k][n]
  __shared__ __align__(32) float  s_new[DD];
  __shared__ float s_red[BLK];
  const int tid = threadIdx.x, lane = tid & 31, wave = tid >> 5;
  const int c0 = tid * 32;                 // this thread's contiguous chunk
  const int kk = tid >> 1, dd0 = (tid & 1) * 32;  // dest-major build coords

  for (int j = 0; j < 32; ++j) s_Al[c0 + j] = (c0 + j == 0) ? 1.f : 0.f;
  __syncthreads();

  float acc = 0.f;
  for (int i = 0; i < 256; ++i) {
    // rescale Al by max-abs, accumulate log
    float lmax = 0.f;
#pragma unroll
    for (int j = 0; j < 32; ++j) lmax = fmaxf(lmax, fabsf(s_Al[c0 + j]));
    const float amax = red_max(lmax, s_red);
    acc += logf(amax);
    const float scale = 1.f / amax;
#pragma unroll
    for (int j8 = 0; j8 < 4; ++j8) {
      v8bf w;
#pragma unroll
      for (int e = 0; e < 8; ++e) {
        const float v = s_Al[c0 + j8 * 8 + e] * scale;
        s_Al[c0 + j8 * 8 + e] = v;
        w[e] = (__bf16)v;
      }
      *(v8bf*)&s_Albf[c0 + j8 * 8] = w;
    }
    __syncthreads();

    const bool outsite = ((i & 1) == 0) || (i == 255);
    const int  nC      = outsite ? 4 : 2;
    if (outsite) {
      const float* Mo = Mo_all + (size_t)((i == 255) ? 128 : (i >> 1)) * (DD * 4);
#pragma unroll
      for (int j8 = 0; j8 < 4; ++j8) {
        v8bf w[4];
#pragma unroll
        for (int e = 0; e < 8; ++e) {
          const int dd = dd0 + j8 * 8 + e;
          const float* m4 = Mo + (size_t)(dd * 64 + kk) * 4;
#pragma unroll
          for (int c4 = 0; c4 < 4; ++c4) w[c4][e] = (__bf16)m4[c4];
        }
#pragma unroll
        for (int c4 = 0; c4 < 4; ++c4)
          *(v8bf*)&s_Mt[c4][kk * 64 + dd0 + j8 * 8] = w[c4];
      }
    } else {
      const float* Mi = Mi_all + (size_t)((i - 1) >> 1) * (DD * 2);
#pragma unroll
      for (int j8 = 0; j8 < 4; ++j8) {
        v8bf w0, w1;
#pragma unroll
        for (int e = 0; e < 8; ++e) {
          const int dd = dd0 + j8 * 8 + e;
          const float* m2 = Mi + (size_t)(dd * 64 + kk) * 2;
          w0[e] = (__bf16)m2[0];
          w1[e] = (__bf16)m2[1];
        }
        *(v8bf*)&s_Mt[0][kk * 64 + dd0 + j8 * 8] = w0;
        *(v8bf*)&s_Mt[1][kk * 64 + dd0 + j8 * 8] = w1;
      }
    }
    __syncthreads();

    // U_c = Al @ M_c
    for (int c4 = 0; c4 < nC; ++c4)
#pragma unroll
      for (int tn = 0; tn < 4; ++tn) {
        v8f c = {};
#pragma unroll
        for (int kt = 0; kt < 2; ++kt)
          c = WMMA(frag_a(s_Albf, wave * 16, kt * 32, lane),
                   frag_bT(s_Mt[c4], kt * 32, tn * 16, lane), c);
        store_T_bf16(s_Ut[c4], wave * 16, tn * 16, lane, c);
      }
    __syncthreads();

    // Al' = sum_c M_c^T @ U_c   (A-op (m,k) = M_c[k][m] = s_Mt[c][m][k])
#pragma unroll
    for (int tn = 0; tn < 4; ++tn) {
      v8f c = {};
      for (int c4 = 0; c4 < nC; ++c4)
#pragma unroll
        for (int kt = 0; kt < 2; ++kt)
          c = WMMA(frag_a(s_Mt[c4], wave * 16, kt * 32, lane),
                   frag_bT(s_Ut[c4], kt * 32, tn * 16, lane), c);
      store_T_f32(s_new, wave * 16, tn * 16, lane, c);   // symmetric: ^T ok
    }
    __syncthreads();
#pragma unroll
    for (int j = 0; j < 32; ++j) s_Al[c0 + j] = s_new[c0 + j];
    __syncthreads();
  }
  if (tid == 0) lognorm[0] = acc + logf(s_Al[0]);
}

// =====================================================================
// Kernel 2: batched forward chain -> out[b] = logp[b] - lognorm
//   Uses Al symmetry:  A_t = P @ X_t ; U_t = Al @ A_t ; Al' = sum_t A_t^T @ U_t
// =====================================================================
__global__ __launch_bounds__(BLK) void chain_kernel(
    const float* __restrict__ inp,      // [256,64,2]
    const float* __restrict__ Mi_all,   // [127,64,64,2]
    const float* __restrict__ Mo_all,   // [129,64,64,2,2]
    const float* __restrict__ lognorm,
    float* __restrict__ out) {
  __shared__ float s_inp[512];
  __shared__ __align__(32) float  s_Al[DD];
  __shared__ __align__(32) __bf16 s_Albf[DD];
  __shared__ __align__(32) __bf16 s_P[DD];        // row-major (A-op only)
  __shared__ __align__(32) __bf16 s_Xt[2][DD];    // ST_X[t][s][r] = X_t[r][s]
  __shared__ __align__(32) __bf16 s_At[2][DD];    // ST_A[t][s][r] = A_t[r][s]
  __shared__ __align__(32) __bf16 s_Ut[2][DD];    // ST_U[t][n][k] = U_t[k][n]
  __shared__ __align__(32) float  s_new[DD];
  __shared__ float s_red[BLK];
  __shared__ float s_vec[BLK];
  const int tid = threadIdx.x, lane = tid & 31, wave = tid >> 5;
  const int b = blockIdx.x;
  const int c0 = tid * 32;
  const int sc = tid >> 1, r0v = (tid & 1) * 32;   // dest-major build coords

  // stage this batch item's input scalars: s_inp[n*2+u] = inp[n,b,u]
  for (int i = tid; i < 512; i += BLK)
    s_inp[i] = inp[(size_t)(i >> 1) * 128 + b * 2 + (i & 1)];
  __syncthreads();

  float acc = 0.f, acc_outS = 0.f;

  // ---- site 0: rank-2 init  Al = a0 a0^T + a1 a1^T, then rescale ----
  {
    const float w0 = s_inp[0], w1 = s_inp[1];
    { const int r = tid >> 1, t = tid & 1;   // A0[r,t] = sum_u w_u * Mo[0][0,r,u,t]
      s_vec[tid] = w0 * Mo_all[r * 4 + t] + w1 * Mo_all[r * 4 + 2 + t]; }
    __syncthreads();
    float lmax = 0.f;
#pragma unroll
    for (int j = 0; j < 32; ++j) {
      const int idx = c0 + j, l = idx >> 6, r = idx & 63;
      const float v = s_vec[l * 2] * s_vec[r * 2] + s_vec[l * 2 + 1] * s_vec[r * 2 + 1];
      s_new[idx] = v; lmax = fmaxf(lmax, fabsf(v));
    }
    const float amax = red_max(lmax, s_red);
    acc += logf(amax);
    const float scale = (amax == 0.f) ? 0.f : 1.f / amax;
#pragma unroll
    for (int j8 = 0; j8 < 4; ++j8) {
      v8bf w;
#pragma unroll
      for (int e = 0; e < 8; ++e) {
        const float v = s_new[c0 + j8 * 8 + e] * scale;
        s_Al[c0 + j8 * 8 + e] = v;
        w[e] = (__bf16)v;
      }
      *(v8bf*)&s_Albf[c0 + j8 * 8] = w;
    }
    __syncthreads();
  }

  // ---- chain steps k = 1..127 (odd site 2k-1 + even site 2k fused) ----
  for (int k = 1; k < 128; ++k) {
    const float* Mi = Mi_all + (size_t)(k - 1) * (DD * 2);
    const float* Mo = Mo_all + (size_t)k * (DD * 4);
    const float wp0 = s_inp[(2 * k - 1) * 2], wp1 = s_inp[(2 * k - 1) * 2 + 1];
    const float wx0 = s_inp[4 * k],           wx1 = s_inp[4 * k + 1];

    // prefetch next site's weights (L2-resident after warmup)
    if (k < 127) {
      __builtin_prefetch(Mi + DD * 2 + tid * 64, 0, 1);
      __builtin_prefetch(Mo + DD * 4 + tid * 64, 0, 1);
      __builtin_prefetch(Mo + DD * 4 + (tid + BLK) * 64, 0, 1);
    }

    // P[l,r] = wp0*Mi[l,r,0] + wp1*Mi[l,r,1] -> amax-normalize -> out_S
    float lmax = 0.f;
#pragma unroll
    for (int j = 0; j < 32; ++j) {
      const int idx = c0 + j;
      const float v = wp0 * Mi[idx * 2] + wp1 * Mi[idx * 2 + 1];
      s_new[idx] = v; lmax = fmaxf(lmax, fabsf(v));
    }
    float amax = red_max(lmax, s_red);
    acc_outS += logf(amax);
    float scale = (amax == 0.f) ? 0.f : 1.f / amax;
#pragma unroll
    for (int j8 = 0; j8 < 4; ++j8) {
      v8bf w;
#pragma unroll
      for (int e = 0; e < 8; ++e) w[e] = (__bf16)(s_new[c0 + j8 * 8 + e] * scale);
      *(v8bf*)&s_P[c0 + j8 * 8] = w;
    }
    // X_t build, transposed destination: ST_X[t][sc][r] = X_t[r][sc]
#pragma unroll
    for (int j8 = 0; j8 < 4; ++j8) {
      v8bf w0, w1;
#pragma unroll
      for (int e = 0; e < 8; ++e) {
        const int r = r0v + j8 * 8 + e;
        const float* m4 = Mo + (size_t)(r * 64 + sc) * 4;   // [r,s,u,t]
        w0[e] = (__bf16)(wx0 * m4[0] + wx1 * m4[2]);
        w1[e] = (__bf16)(wx0 * m4[1] + wx1 * m4[3]);
      }
      *(v8bf*)&s_Xt[0][sc * 64 + r0v + j8 * 8] = w0;
      *(v8bf*)&s_Xt[1][sc * 64 + r0v + j8 * 8] = w1;
    }
    __syncthreads();

    // A_t = P @ X_t  (store transposed)
#pragma unroll
    for (int t = 0; t < 2; ++t)
#pragma unroll
      for (int tn = 0; tn < 4; ++tn) {
        v8f c = {};
#pragma unroll
        for (int kt = 0; kt < 2; ++kt)
          c = WMMA(frag_a(s_P, wave * 16, kt * 32, lane),
                   frag_bT(s_Xt[t], kt * 32, tn * 16, lane), c);
        store_T_bf16(s_At[t], wave * 16, tn * 16, lane, c);
      }
    __syncthreads();

    // U_t = Al @ A_t  (Al symmetric; store transposed)
#pragma unroll
    for (int t = 0; t < 2; ++t)
#pragma unroll
      for (int tn = 0; tn < 4; ++tn) {
        v8f c = {};
#pragma unroll
        for (int kt = 0; kt < 2; ++kt)
          c = WMMA(frag_a(s_Albf, wave * 16, kt * 32, lane),
                   frag_bT(s_At[t], kt * 32, tn * 16, lane), c);
        store_T_bf16(s_Ut[t], wave * 16, tn * 16, lane, c);
      }
    __syncthreads();

    // Al' = sum_t A_t^T @ U_t  (A-op (m,k) = A_t[k][m] = s_At[t][m][k])
#pragma unroll
    for (int tn = 0; tn < 4; ++tn) {
      v8f c = {};
#pragma unroll
      for (int t = 0; t < 2; ++t)
#pragma unroll
        for (int kt = 0; kt < 2; ++kt)
          c = WMMA(frag_a(s_At[t], wave * 16, kt * 32, lane),
                   frag_bT(s_Ut[t], kt * 32, tn * 16, lane), c);
      store_T_f32(s_new, wave * 16, tn * 16, lane, c);    // symmetric: ^T ok
    }
    __syncthreads();

    // rescale
    lmax = 0.f;
#pragma unroll
    for (int j = 0; j < 32; ++j) lmax = fmaxf(lmax, fabsf(s_new[c0 + j]));
    amax = red_max(lmax, s_red);
    acc += logf(amax);
    scale = (amax == 0.f) ? 0.f : 1.f / amax;
#pragma unroll
    for (int j8 = 0; j8 < 4; ++j8) {
      v8bf w;
#pragma unroll
      for (int e = 0; e < 8; ++e) {
        const float v = s_new[c0 + j8 * 8 + e] * scale;
        s_Al[c0 + j8 * 8 + e] = v;
        w[e] = (__bf16)v;
      }
      *(v8bf*)&s_Albf[c0 + j8 * 8] = w;
    }
    __syncthreads();
  }

  // ---- final site 255: rank-2 close of the chain ----
  {
    const float w0 = s_inp[510], w1 = s_inp[511];
    const float* MoL = Mo_all + (size_t)128 * (DD * 4);
    { const int dn = tid >> 1, t = tid & 1;  // Af[dn,t] = sum_u w_u * MoL[dn,0,u,t]
      s_vec[tid] = w0 * MoL[dn * 256 + t] + w1 * MoL[dn * 256 + 2 + t]; }
    __syncthreads();
    const int u = tid >> 1, t = tid & 1;     // T[u,t] = sum_dn Al[dn,u]*Af[dn,t]
    float tv = 0.f;
    for (int dn = 0; dn < 64; ++dn) tv += s_Al[dn * 64 + u] * s_vec[dn * 2 + t];
    const float val = red_sum(tv * s_vec[tid], s_red);  // Al'[0,0]
    acc += logf(fabsf(val));                            // |Al00| after norm == 1
    if (tid == 0) out[b] = acc + 2.f * acc_outS - lognorm[0];
  }
}

extern "C" void kernel_launch(void* const* d_in, const int* in_sizes, int n_in,
                              void* d_out, int out_size, void* d_ws, size_t ws_size,
                              hipStream_t stream) {
  (void)in_sizes; (void)n_in; (void)out_size; (void)ws_size;
  const float* inp        = (const float*)d_in[0];
  const float* mps_input  = (const float*)d_in[1];
  const float* mps_output = (const float*)d_in[2];
  float* lognorm = (float*)d_ws;
  float* out     = (float*)d_out;

  lognorm_kernel<<<1, BLK, 0, stream>>>(mps_input, mps_output, lognorm);
  chain_kernel<<<64, BLK, 0, stream>>>(inp, mps_input, mps_output, lognorm, out);
}
